// RWKV_89928025244447
// MI455X (gfx1250) — matrix operand order repaired and checked by reading, not compiled, not run on hardware
//
#include <hip/hip_runtime.h>
#include <hip/hip_bf16.h>

typedef __bf16 bf16;
typedef __attribute__((ext_vector_type(16))) __bf16 v16bf;
typedef __attribute__((ext_vector_type(8)))  float  v8f;

#define LDSS 40  // 32 + 8 pad (bf16 units): row stride 80B -> conflict-free

// ---------------------------------------------------------------------------
// CDNA5 async global->LDS copy (no VGPR staging; tracked by ASYNCcnt).
// GVS addressing: mem = SGPR64 base + VGPR u32 byte offset. First operand is
// the per-lane LDS byte address (low 32 bits of the generic shared pointer).
// ---------------------------------------------------------------------------
__device__ __forceinline__ void async_copy_b128(unsigned lds_addr, const void* sbase,
                                                unsigned voff_bytes) {
  asm volatile("global_load_async_to_lds_b128 %0, %1, %2"
               :
               : "v"(lds_addr), "v"(voff_bytes), "s"(sbase)
               : "memory");
}
__device__ __forceinline__ void wait_async0() {
  asm volatile("s_wait_asynccnt 0x0" ::: "memory");
}

// ---------------------------------------------------------------------------
// f32 -> bf16 elementwise convert
// ---------------------------------------------------------------------------
__global__ void cvt_f32_bf16(const float* __restrict__ src, bf16* __restrict__ dst,
                             size_t n) {
  size_t i = (size_t)blockIdx.x * blockDim.x + threadIdx.x;
  if (i < n) dst[i] = (bf16)src[i];
}

// ---------------------------------------------------------------------------
// f32 [R][C] -> bf16 transposed [C][R], LDS-tiled
// ---------------------------------------------------------------------------
__global__ void cvt_transpose_bf16(const float* __restrict__ src, bf16* __restrict__ dst,
                                   int R, int C) {
  __shared__ float tile[32][33];
  int c0 = blockIdx.x * 32, r0 = blockIdx.y * 32;
  int tx = threadIdx.x, ty = threadIdx.y;  // blockDim (32,8)
  for (int i = ty; i < 32; i += 8)
    if (r0 + i < R && c0 + tx < C)
      tile[i][tx] = src[(size_t)(r0 + i) * C + c0 + tx];
  __syncthreads();
  for (int i = ty; i < 32; i += 8)
    if (c0 + i < C && r0 + tx < R)
      dst[(size_t)(c0 + i) * R + r0 + tx] = (bf16)tile[tx][i];
}

// ---------------------------------------------------------------------------
// bf16 WMMA GEMM: C[M][N] = A[M][K] * Bt[N][K]^T + bias[N]   (f32 accumulate)
// Block tile 128x128x32, 256 threads = 8 waves, each wave 32x64 (2x4 tiles).
// Global->LDS via async copies, double-buffered; no staging VGPRs.
// ---------------------------------------------------------------------------
__global__ __launch_bounds__(256)
void wmma_gemm_bf16(const bf16* __restrict__ A, const bf16* __restrict__ Bt,
                    const float* __restrict__ bias, float* __restrict__ C,
                    int M, int N, int K) {
  __shared__ __align__(16) bf16 sA[2][128 * LDSS];
  __shared__ __align__(16) bf16 sB[2][128 * LDSS];

  const int tid  = threadIdx.x;
  const int wave = tid >> 5;
  const int lane = tid & 31;
  const int wr   = wave >> 1;   // 0..3 -> wave row (32 rows each)
  const int wc   = wave & 1;    // 0..1 -> wave col (64 cols each)
  const int g    = lane >> 4;   // half-wave
  const int l15  = lane & 15;

  const int rowBase = blockIdx.y * 128;
  const int colBase = blockIdx.x * 128;

  // Issue one 128x32 bf16 tile for A and B each: 512 x 16B chunks, 2 per thread.
  auto issue_tile = [&](int kk, int buf) {
#pragma unroll
    for (int j = 0; j < 2; ++j) {
      int e = tid + j * 256;           // [0,512)
      int r = e >> 2;                  // 0..127
      int c = (e & 3) << 3;            // 0,8,16,24
      unsigned aoff = (unsigned)(((size_t)(rowBase + r) * K + kk + c) * sizeof(bf16));
      unsigned boff = (unsigned)(((size_t)(colBase + r) * K + kk + c) * sizeof(bf16));
      async_copy_b128((unsigned)(uintptr_t)&sA[buf][r * LDSS + c], A, aoff);
      async_copy_b128((unsigned)(uintptr_t)&sB[buf][r * LDSS + c], Bt, boff);
    }
  };

  v8f acc[2][4];
#pragma unroll
  for (int mi = 0; mi < 2; ++mi)
#pragma unroll
    for (int ni = 0; ni < 4; ++ni)
#pragma unroll
      for (int e = 0; e < 8; ++e) acc[mi][ni][e] = 0.0f;

  issue_tile(0, 0);

  const int nk = K >> 5;
  for (int kt = 0; kt < nk; ++kt) {
    const int buf = kt & 1;

    wait_async0();        // this wave's async writes for tile kt are in LDS
    __syncthreads();      // all waves' writes visible; all done reading buf^1

    if (kt + 1 < nk) issue_tile((kt + 1) << 5, buf ^ 1);

    // Fragment loads follow the CDNA5 WMMA VGPR layouts (ISA 7.12.2):
    // A (16x32 bf16): lane half g, elements 0..7 <-> k=8g..8g+7, 8..15 <-> k=16+8g..
    // B (32x16 bf16, stored transposed): same k order as A (0..7 <-> k=8g.., 8..15 <-> k=16+8g..)
    union Frag { v16bf v; uint4 q[2]; };
    Frag fa[2], fb[4];
#pragma unroll
    for (int mi = 0; mi < 2; ++mi) {
      const bf16* rp = &sA[buf][(wr * 32 + mi * 16 + l15) * LDSS];
      fa[mi].q[0] = *reinterpret_cast<const uint4*>(rp + g * 8);
      fa[mi].q[1] = *reinterpret_cast<const uint4*>(rp + g * 8 + 16);
    }
#pragma unroll
    for (int ni = 0; ni < 4; ++ni) {
      const bf16* rp = &sB[buf][(wc * 64 + ni * 16 + l15) * LDSS];
      fb[ni].q[0] = *reinterpret_cast<const uint4*>(rp + g * 8);
      fb[ni].q[1] = *reinterpret_cast<const uint4*>(rp + g * 8 + 16);
    }

#pragma unroll
    for (int mi = 0; mi < 2; ++mi)
#pragma unroll
      for (int ni = 0; ni < 4; ++ni)
        acc[mi][ni] = __builtin_amdgcn_wmma_f32_16x16x32_bf16(
            false, fa[mi].v, false, fb[ni].v, (short)0, acc[mi][ni], false, false);
  }

  // Epilogue: C/D layout — VGPR v holds row v + 8g, lane&15 = column.
#pragma unroll
  for (int ni = 0; ni < 4; ++ni) {
    const int col = colBase + wc * 64 + ni * 16 + l15;
    const float bv = bias[col];
#pragma unroll
    for (int mi = 0; mi < 2; ++mi) {
      const int row0 = rowBase + wr * 32 + mi * 16 + g * 8;
#pragma unroll
      for (int v = 0; v < 8; ++v)
        C[(size_t)(row0 + v) * N + col] = acc[mi][ni][v] + bv;
    }
  }
}

// ---------------------------------------------------------------------------
// Linear recurrence y_t = lam*y_{t-1} + u_t over L=4096, split into 32 chunks
// of 128. B=4, U=1024, rkv layout [B*L][3072] with r|k|v column blocks.
// ---------------------------------------------------------------------------
__global__ void scan_pass1(const float* __restrict__ rkv, const float* __restrict__ nu_log,
                           float* __restrict__ end1, float* __restrict__ end2) {
  int idx = blockIdx.x * blockDim.x + threadIdx.x;  // B*32*U = 131072
  int u = idx & 1023;
  int c = (idx >> 10) & 31;
  int b = idx >> 15;
  float lam = __expf(-__expf(nu_log[u]));
  float y1 = 0.f, y2 = 0.f;
  const float* base = rkv + ((size_t)b * 4096 + (size_t)c * 128) * 3072;
  for (int i = 0; i < 128; ++i) {
    if (i + 8 < 128) {
      __builtin_prefetch(&base[(size_t)(i + 8) * 3072 + 1024 + u], 0, 0);
      __builtin_prefetch(&base[(size_t)(i + 8) * 3072 + 2048 + u], 0, 0);
    }
    float kr = base[(size_t)i * 3072 + 1024 + u];
    float vr = base[(size_t)i * 3072 + 2048 + u];
    float k  = __expf(kr);
    float kv = k * vr;
    y1 = fmaf(lam, y1, kv);
    y2 = fmaf(lam, y2, k);
  }
  end1[idx] = y1;
  end2[idx] = y2;
}

__global__ void scan_pass2(const float* __restrict__ end1, const float* __restrict__ end2,
                           const float* __restrict__ nu_log,
                           float* __restrict__ car1, float* __restrict__ car2) {
  int idx = blockIdx.x * blockDim.x + threadIdx.x;  // B*U = 4096
  int u = idx & 1023;
  int b = idx >> 10;
  float nu   = __expf(nu_log[u]);
  float lamP = __expf(-nu * 128.0f);  // lam^128
  float c1 = 0.f, c2 = 0.f;
  for (int c = 0; c < 32; ++c) {
    size_t o = ((size_t)(b * 32 + c) << 10) + u;
    car1[o] = c1;
    car2[o] = c2;
    c1 = fmaf(lamP, c1, end1[o]);
    c2 = fmaf(lamP, c2, end2[o]);
  }
}

__global__ void scan_pass3(const float* __restrict__ rkv,
                           const float* __restrict__ car1, const float* __restrict__ car2,
                           const float* __restrict__ nu_log, const float* __restrict__ gamma_log,
                           bf16* __restrict__ xbf) {
  int idx = blockIdx.x * blockDim.x + threadIdx.x;  // B*32*U
  int u = idx & 1023;
  int c = (idx >> 10) & 31;
  int b = idx >> 15;
  float nu    = __expf(nu_log[u]);
  float lam   = __expf(-nu);
  float gamma = __expf(nu_log[u] + gamma_log[u]) - 1.0f;
  float y1 = car1[idx], y2 = car2[idx];
  const float* base = rkv + ((size_t)b * 4096 + (size_t)c * 128) * 3072;
  bf16* xb = xbf + ((size_t)b * 4096 + (size_t)c * 128) * 1024;
  for (int i = 0; i < 128; ++i) {
    if (i + 8 < 128) {
      __builtin_prefetch(&base[(size_t)(i + 8) * 3072 + u], 0, 0);
      __builtin_prefetch(&base[(size_t)(i + 8) * 3072 + 1024 + u], 0, 0);
      __builtin_prefetch(&base[(size_t)(i + 8) * 3072 + 2048 + u], 0, 0);
    }
    float rr = base[(size_t)i * 3072 + u];
    float kr = base[(size_t)i * 3072 + 1024 + u];
    float vr = base[(size_t)i * 3072 + 2048 + u];
    float k  = __expf(kr);
    float kv = k * vr;
    y1 = fmaf(lam, y1, kv);
    y2 = fmaf(lam, y2, k);
    float r = 1.0f / (1.0f + __expf(-rr));
    float x = (y1 + gamma * kv) / (y2 + gamma * k + 1e-6f) * r;
    xb[(size_t)i * 1024 + u] = (bf16)x;
  }
}

// ---------------------------------------------------------------------------
extern "C" void kernel_launch(void* const* d_in, const int* in_sizes, int n_in,
                              void* d_out, int out_size, void* d_ws, size_t ws_size,
                              hipStream_t stream) {
  const float* x_in   = (const float*)d_in[0];  // [B,L,H]
  const float* W_rkv  = (const float*)d_in[1];  // [H,3U]
  const float* b_rkv  = (const float*)d_in[2];  // [3U]
  const float* W_o    = (const float*)d_in[3];  // [U,H]
  const float* b_o    = (const float*)d_in[4];  // [H]
  const float* nu_log = (const float*)d_in[5];  // [U]
  const float* gl     = (const float*)d_in[6];  // [U]
  float* out = (float*)d_out;                   // [B,L,H] f32

  const int Bsz = 4, L = 4096, H = 1024, U = 1024;
  const size_t M  = (size_t)Bsz * L;  // 16384
  const size_t N1 = 3 * U;            // 3072

  char* p = (char*)d_ws;
  auto wsalloc = [&](size_t bytes) {
    char* q = p;
    p += (bytes + 255) & ~(size_t)255;
    return (void*)q;
  };
  bf16*  Xbf  = (bf16*) wsalloc(M * H * 2);          //  32 MB
  bf16*  W1t  = (bf16*) wsalloc(N1 * H * 2);         //   6 MB  [3U][H]
  bf16*  W2t  = (bf16*) wsalloc((size_t)H * U * 2);  //   2 MB  [H][U]
  float* rkv  = (float*)wsalloc(M * N1 * 4);         // 192 MB
  bf16*  xbf  = (bf16*) wsalloc(M * U * 2);          //  32 MB
  float* end1 = (float*)wsalloc((size_t)Bsz * 32 * U * 4);
  float* end2 = (float*)wsalloc((size_t)Bsz * 32 * U * 4);
  float* car1 = (float*)wsalloc((size_t)Bsz * 32 * U * 4);
  float* car2 = (float*)wsalloc((size_t)Bsz * 32 * U * 4);

  // 1) precision staging
  cvt_f32_bf16<<<(unsigned)((M * H) / 256), 256, 0, stream>>>(x_in, Xbf, M * H);
  cvt_transpose_bf16<<<dim3(N1 / 32, H / 32), dim3(32, 8), 0, stream>>>(W_rkv, W1t, H, (int)N1);
  cvt_transpose_bf16<<<dim3(H / 32, U / 32), dim3(32, 8), 0, stream>>>(W_o, W2t, U, H);

  // 2) rkv = X @ W_rkv + b_rkv
  wmma_gemm_bf16<<<dim3((unsigned)(N1 / 128), (unsigned)(M / 128)), 256, 0, stream>>>(
      Xbf, W1t, b_rkv, rkv, (int)M, (int)N1, H);

  // 3) chunked parallel scan + gated epilogue -> x (bf16)
  scan_pass1<<<(Bsz * 32 * U) / 256, 256, 0, stream>>>(rkv, nu_log, end1, end2);
  scan_pass2<<<(Bsz * U) / 256, 256, 0, stream>>>(end1, end2, nu_log, car1, car2);
  scan_pass3<<<(Bsz * 32 * U) / 256, 256, 0, stream>>>(rkv, car1, car2, nu_log, gl, xbf);

  // 4) out = x @ W_o + b_o
  wmma_gemm_bf16<<<dim3((unsigned)(H / 128), (unsigned)(M / 128)), 256, 0, stream>>>(
      xbf, W2t, b_o, out, (int)M, H, U);
}
